// Codebook_274877907244
// MI455X (gfx1250) — matrix operand order repaired and checked by reading, not compiled, not run on hardware
//
#include <hip/hip_runtime.h>
#include <hip/hip_bf16.h>

// ---------------- problem constants ----------------
#define KCB   4096            // codebook entries
#define DD    256             // latent dim
#define BB    16              // batch
#define HW    4096            // 64*64
#define NN    (BB * HW)       // 65536 latent vectors
#define MROWS 128             // rows of z per block
#define APAD  264             // padded LDS row stride (bf16 elems); 264*2B = 16B-aligned
#define NT    (KCB / 16)      // 256 column tiles
#define ZQ_SIZE (BB * DD * HW)   // 16777216

typedef __attribute__((ext_vector_type(16))) __bf16 v16bf;
typedef __attribute__((ext_vector_type(8)))  __bf16 v8bf;
typedef __attribute__((ext_vector_type(8)))  float  v8f;
typedef __attribute__((ext_vector_type(4)))  int    v4i;

// builtin param types: (v4i addrspace(1)*, v4i addrspace(3)*, imm int, imm int)
typedef __attribute__((address_space(1))) v4i gv4i;
typedef __attribute__((address_space(3))) v4i sv4i;

#if __has_builtin(__builtin_amdgcn_global_load_async_to_lds_b128)
#define HAVE_ASYNC_LDS 1
#else
#define HAVE_ASYNC_LDS 0
#endif

__device__ __forceinline__ void async_wait0() {
#if __has_builtin(__builtin_amdgcn_s_wait_asynccnt)
    __builtin_amdgcn_s_wait_asynccnt(0);
#elif HAVE_ASYNC_LDS
    asm volatile("s_wait_asynccnt 0x0" ::: "memory");
#endif
}

__device__ __forceinline__ unsigned short f2bf(float f) {
    unsigned u = __float_as_uint(f);
    u += 0x7FFFu + ((u >> 16) & 1u);   // round-to-nearest-even
    return (unsigned short)(u >> 16);
}

// ---------------- kernel 1: emb -> bf16 + ||e||^2 ----------------
__global__ __launch_bounds__(256) void vq_prep(const float* __restrict__ emb,
                                               unsigned short* __restrict__ embbf,
                                               float* __restrict__ enorm) {
    __shared__ float red[256];
    const int k = blockIdx.x;
    const int t = threadIdx.x;
    float v = emb[k * DD + t];
    embbf[k * DD + t] = f2bf(v);
    red[t] = v * v;
    __syncthreads();
    for (int s = 128; s > 0; s >>= 1) {
        if (t < s) red[t] += red[t + s];
        __syncthreads();
    }
    if (t == 0) enorm[k] = red[0];
}

// ---------------- kernel 2: distances via WMMA + argmin + gather + loss ----------------
__global__ __launch_bounds__(256) void vq_main(const float* __restrict__ z,
                                               const float* __restrict__ emb,
                                               const unsigned short* __restrict__ embbf,
                                               const float* __restrict__ enorm,
                                               float* __restrict__ out_zq,
                                               float* __restrict__ out_idx,
                                               float* __restrict__ partials) {
    __shared__ unsigned short a_lds[MROWS * APAD];    // 128 z-rows, bf16, padded (~67.5 KB)
    __shared__ unsigned short b_lds[2][16 * APAD];    // double-buffered 16-row emb tiles
    __shared__ int   idx_lds[MROWS];
    __shared__ float red[256];

    const int tid  = threadIdx.x;
    const int lane = tid & 31;
    const int wave = tid >> 5;
    const int rowBase = blockIdx.x * MROWS;
    const int bidx = rowBase >> 12;          // batch index (HW = 4096 rows per image)
    const int hw0  = rowBase & (HW - 1);

    // ---- stage A tile: z[rowBase .. rowBase+127] x 256 ch -> bf16 LDS ----
    {
        const int nl    = tid & 127;         // local row
        const int chalf = tid >> 7;          // 0 / 1
        #pragma unroll 4
        for (int i = 0; i < 128; ++i) {
            const int c = i * 2 + chalf;
            const float v = z[((size_t)(bidx * DD + c)) * HW + hw0 + nl];
            a_lds[nl * APAD + c] = f2bf(v);
        }
    }

    // per-thread B-staging geometry: 512 uint4 per 8KB tile, 2 per thread
    // LDS row = 33 uint4 (APAD/8), source row = 32 uint4
    const int u4a  = tid;            // first uint4 id
    const int u4b  = tid + 256;      // second uint4 id
    const int dsta = ((u4a >> 5) * 33 + (u4a & 31));   // uint4 index into b_lds buffer
    const int dstb = ((u4b >> 5) * 33 + (u4b & 31));

#if HAVE_ASYNC_LDS
    #define STAGE_TILE(ct, buf)                                                          \
        do {                                                                             \
            const char* _s = (const char*)embbf + (size_t)(ct) * 8192;                   \
            char* _d = (char*)&b_lds[(buf)][0];                                          \
            __builtin_amdgcn_global_load_async_to_lds_b128(                              \
                (gv4i*)(_s + (size_t)u4a * 16),                                          \
                (sv4i*)(_d + (size_t)dsta * 16), 0, 0);                                  \
            __builtin_amdgcn_global_load_async_to_lds_b128(                              \
                (gv4i*)(_s + (size_t)u4b * 16),                                          \
                (sv4i*)(_d + (size_t)dstb * 16), 0, 0);                                  \
        } while (0)
#else
    uint4 rg0, rg1;                   // register staging for fallback path
    #define LOAD_TILE_REGS(ct)                                                           \
        do {                                                                             \
            const uint4* _s = (const uint4*)((const char*)embbf + (size_t)(ct) * 8192);  \
            rg0 = _s[u4a];                                                               \
            rg1 = _s[u4b];                                                               \
        } while (0)
    #define STORE_TILE_REGS(buf)                                                         \
        do {                                                                             \
            uint4* _d = (uint4*)&b_lds[(buf)][0];                                        \
            _d[dsta] = rg0;                                                              \
            _d[dstb] = rg1;                                                              \
        } while (0)
#endif

    // prologue: bring tile 0 into buffer 0 (and finish the A staging barrier)
#if HAVE_ASYNC_LDS
    STAGE_TILE(0, 0);
    async_wait0();
#else
    LOAD_TILE_REGS(0);
    STORE_TILE_REGS(0);
#endif
    __syncthreads();

    // ---- per-wave A fragments: rows wave*16 + (lane&15), all 256 channels ----
    // ISA 16-bit A layout: lanes 0-15 hold K {c0..c0+7, c0+16..c0+23}; lanes 16-31 hold +8.
    const int arow  = wave * 16 + (lane & 15);
    const int ahalf = (lane >> 4) * 8;
    v16bf afr[8];
    #pragma unroll
    for (int ks = 0; ks < 8; ++ks) {
        const int c0 = ks * 32;
        v8bf lo = *reinterpret_cast<const v8bf*>(&a_lds[arow * APAD + c0 + ahalf]);
        v8bf hi = *reinterpret_cast<const v8bf*>(&a_lds[arow * APAD + c0 + 16 + ahalf]);
        afr[ks] = __builtin_shufflevector(lo, hi, 0, 1, 2, 3, 4, 5, 6, 7,
                                                  8, 9, 10, 11, 12, 13, 14, 15);
    }

    float mval[8];
    int   midx[8];
    #pragma unroll
    for (int r = 0; r < 8; ++r) { mval[r] = 3.402823466e+38f; midx[r] = 0; }

    // ISA 16-bit B layout: lane = column (N), lanes 0-15 K {c0..c0+7, c0+16..c0+23}; lanes 16-31 hold +8.
    const int bcol  = lane & 15;
    const int bhalf = (lane >> 4) * 8;

    #pragma unroll 2
    for (int ct = 0; ct < NT; ++ct) {
        const int cur = ct & 1;
        // kick off the next tile's fetch so it hides under the WMMA stream
#if HAVE_ASYNC_LDS
        if (ct + 1 < NT) STAGE_TILE(ct + 1, cur ^ 1);
#else
        if (ct + 1 < NT) LOAD_TILE_REGS(ct + 1);
#endif

        const unsigned short* bl = &b_lds[cur][0];
        v8f acc = {};
        #pragma unroll
        for (int ks = 0; ks < 8; ++ks) {
            const int c0 = ks * 32;
            v8bf lo = *reinterpret_cast<const v8bf*>(&bl[bcol * APAD + c0 + bhalf]);
            v8bf hi = *reinterpret_cast<const v8bf*>(&bl[bcol * APAD + c0 + bhalf + 16]);
            v16bf bfr = __builtin_shufflevector(lo, hi, 0, 1, 2, 3, 4, 5, 6, 7,
                                                        8, 9, 10, 11, 12, 13, 14, 15);
            acc = __builtin_amdgcn_wmma_f32_16x16x32_bf16(
                false, afr[ks], false, bfr, (short)0, acc, false, false);
        }

        // d = ||e||^2 - 2 z.e  (||z||^2 constant per row -> irrelevant for argmin)
        const int col = ct * 16 + bcol;
        const float en = enorm[col];
        #pragma unroll
        for (int r = 0; r < 8; ++r) {
            const float d = __builtin_fmaf(-2.0f, acc[r], en);
            if (d < mval[r]) { mval[r] = d; midx[r] = col; }
        }

#if HAVE_ASYNC_LDS
        async_wait0();                 // next tile resident in LDS
#else
        if (ct + 1 < NT) STORE_TILE_REGS(cur ^ 1);
#endif
        __syncthreads();
    }

    // ---- reduce argmin across the 16 column-lanes (xor tree stays in 16-group) ----
    #pragma unroll
    for (int r = 0; r < 8; ++r) {
        float v  = mval[r];
        int   ix = midx[r];
        #pragma unroll
        for (int off = 8; off >= 1; off >>= 1) {
            const float ov = __shfl_xor(v, off, 32);
            const int   oi = __shfl_xor(ix, off, 32);
            if (ov < v || (ov == v && oi < ix)) { v = ov; ix = oi; }
        }
        mval[r] = v; midx[r] = ix;
    }
    // C/D layout: vgpr r, lanes 0-15 -> M=r, lanes 16-31 -> M=8+r
    if ((lane & 15) == 0) {
        const int rbase = wave * 16 + (lane >> 4) * 8;
        #pragma unroll
        for (int r = 0; r < 8; ++r) idx_lds[rbase + r] = midx[r];
    }
    __syncthreads();

    // ---- gather z_q, write coalesced (lane <-> hw), accumulate loss ----
    float lsum = 0.0f;
    {
        const int nl    = tid & 127;
        const int chalf = tid >> 7;
        const int q = idx_lds[nl];
        const float* __restrict__ erow = emb + (size_t)q * DD;
        #pragma unroll 4
        for (int i = 0; i < 128; ++i) {
            const int c = i * 2 + chalf;
            const size_t off = ((size_t)(bidx * DD + c)) * HW + hw0 + nl;
            const float e  = erow[c];
            const float zv = z[off];
            out_zq[off] = e;                       // straight-through forward value
            const float df = e - zv;
            lsum = __builtin_fmaf(df, df, lsum);
        }
        if (tid < 128) out_idx[rowBase + tid] = (float)idx_lds[tid];
    }

    red[tid] = lsum;
    __syncthreads();
    for (int s = 128; s > 0; s >>= 1) {
        if (tid < s) red[tid] += red[tid + s];
        __syncthreads();
    }
    if (tid == 0) partials[blockIdx.x] = red[0];
}

// ---------------- kernel 3: deterministic loss reduction ----------------
__global__ __launch_bounds__(512) void vq_finish(const float* __restrict__ partials,
                                                 float* __restrict__ out_loss) {
    __shared__ float red[512];
    const int t = threadIdx.x;
    red[t] = partials[t];
    __syncthreads();
    for (int s = 256; s > 0; s >>= 1) {
        if (t < s) red[t] += red[t + s];
        __syncthreads();
    }
    // loss = (1 + BETA) * mean((zq - z)^2), BETA = 0.25, N*C = 16777216
    if (t == 0) out_loss[0] = red[0] * (1.25f / 16777216.0f);
}

// ---------------- launch ----------------
extern "C" void kernel_launch(void* const* d_in, const int* in_sizes, int n_in,
                              void* d_out, int out_size, void* d_ws, size_t ws_size,
                              hipStream_t stream) {
    const float* z   = (const float*)d_in[0];   // [16,256,64,64]
    const float* emb = (const float*)d_in[1];   // [4096,256]

    unsigned short* embbf = (unsigned short*)d_ws;                       // 2 MB
    float* enorm    = (float*)((char*)d_ws + (size_t)KCB * DD * 2);      // 16 KB
    float* partials = (float*)((char*)d_ws + (size_t)KCB * DD * 2 + KCB * 4); // 2 KB

    float* out      = (float*)d_out;
    float* out_zq   = out;                      // 16777216 floats
    float* out_idx  = out + ZQ_SIZE;            // 65536 (argmin indices as float)
    float* out_loss = out + ZQ_SIZE + NN;       // 1

    vq_prep<<<KCB, 256, 0, stream>>>(emb, embbf, enorm);
    vq_main<<<NN / MROWS, 256, 0, stream>>>(z, emb, embbf, enorm,
                                            out_zq, out_idx, partials);
    vq_finish<<<1, 512, 0, stream>>>(partials, out_loss);
}
